// HiResPrecipNet_1563368096251
// MI455X (gfx1250) — matrix operand order repaired and checked by reading, not compiled, not run on hardware
//
#include <hip/hip_runtime.h>
#include <hip/hip_bf16.h>

typedef __attribute__((ext_vector_type(16))) _Float16 v16h;
typedef __attribute__((ext_vector_type(8)))  float    v8f;

// ---------- helpers ----------
__device__ __forceinline__ unsigned encOrder(float f) {
  unsigned u = __float_as_uint(f);
  return (u & 0x80000000u) ? ~u : (u | 0x80000000u);
}
__device__ __forceinline__ float decOrder(unsigned u) {
  unsigned b = (u & 0x80000000u) ? (u ^ 0x80000000u) : ~u;
  return __uint_as_float(b);
}

__global__ void k_zero(float* p, long n) {
  long i = (long)blockIdx.x * blockDim.x + threadIdx.x;
  long st = (long)gridDim.x * blockDim.x;
  for (; i < n; i += st) p[i] = 0.f;
}

// ---------- weight convert: Wt[N][Kpad] f16 = transpose(W[K,N]), zero-fill k>=K ----------
__global__ void k_convw(const float* __restrict__ W, int K, int N, int Kpad,
                        _Float16* __restrict__ Wt) {
  int i = blockIdx.x * blockDim.x + threadIdx.x;
  if (i >= N * Kpad) return;
  int n = i / Kpad, k = i - n * Kpad;
  Wt[i] = (_Float16)((k < K) ? W[(long)k * N + n] : 0.f);
}

// ---------- pad-copy: dst[M,ldd] = src[M,K] zero-padded ----------
__global__ void k_pad(const float* __restrict__ src, int K, float* __restrict__ dst,
                      int ldd, long M) {
  long i = (long)blockIdx.x * blockDim.x + threadIdx.x;
  long st = (long)gridDim.x * blockDim.x;
  long tot = M * ldd;
  for (; i < tot; i += st) {
    long n = i / ldd; int k = (int)(i - n * ldd);
    dst[i] = (k < K) ? src[n * K + k] : 0.f;
  }
}

// ---------- WMMA GEMM: C[M,N] = A[M,Kpad] (ld=lda) @ Wt^T + bias, opt relu ----------
// Requirements: M%16==0, N%32==0, Kpad%32==0, lda%4==0, Wt is [N][Kpad] f16.
// One wave computes a 16x32 tile (two accumulators share the A fragment).
__global__ void k_gemm_wmma(const float* __restrict__ A, int lda,
                            const _Float16* __restrict__ Wt,
                            const float* __restrict__ bias,
                            float* __restrict__ Cq, int ldc, int cofs,
                            int M, int N, int Kpad, int relu) {
  int lane = threadIdx.x & 31;
  int wv   = threadIdx.x >> 5;
  int tm = (blockIdx.x * 4 + wv) * 16;
  if (tm >= M) return;                     // wave-uniform exit
  int tn = blockIdx.y * 32;
  int hl  = lane >> 4;
  int l15 = lane & 15;
  // A layout (ISA 7.12.2): lane half hl, row l15; halves i<8 -> k=hl*8+i ; i>=8 -> k=16+hl*8+(i-8)
  const float* pa = A + (long)(tm + l15) * lda + (hl << 3);
  // B layout: col = l15 (+tn), i<8 -> k=hl*8+i ; i>=8 -> k=16+hl*8+(i-8) (matches A)
  const _Float16* pb0 = Wt + (long)(tn + l15) * Kpad + (hl << 3);
  const _Float16* pb1 = pb0 + (long)16 * Kpad;
  v8f acc0 = {}, acc1 = {};
  for (int kk = 0; kk < Kpad; kk += 32) {
    union { float4 q[4]; float f[16]; } af;
    af.q[0] = *(const float4*)(pa + kk);
    af.q[1] = *(const float4*)(pa + kk + 4);
    af.q[2] = *(const float4*)(pa + kk + 16);
    af.q[3] = *(const float4*)(pa + kk + 20);
    union { v16h v; _Float16 e[16]; } a;
#pragma unroll
    for (int i = 0; i < 16; ++i) a.e[i] = (_Float16)af.f[i];
    union { v16h v; float4 q[2]; } b0, b1;
    b0.q[0] = *(const float4*)(pb0 + kk);
    b0.q[1] = *(const float4*)(pb0 + kk + 16);
    b1.q[0] = *(const float4*)(pb1 + kk);
    b1.q[1] = *(const float4*)(pb1 + kk + 16);
    acc0 = __builtin_amdgcn_wmma_f32_16x16x32_f16(false, a.v, false, b0.v,
                                                  (short)0, acc0, false, false);
    acc1 = __builtin_amdgcn_wmma_f32_16x16x32_f16(false, a.v, false, b1.v,
                                                  (short)0, acc1, false, false);
  }
  // C/D layout: vgpr r -> row = hl*8+r, col = l15
  int rbase = tm + (hl << 3);
  int c0 = tn + l15;
  int c1 = c0 + 16;
  float bv0 = bias ? bias[c0] : 0.f;
  float bv1 = bias ? bias[c1] : 0.f;
#pragma unroll
  for (int r = 0; r < 8; ++r) {
    long rr = rbase + r;
    float v0 = acc0[r] + bv0;
    float v1 = acc1[r] + bv1;
    if (relu) { v0 = fmaxf(v0, 0.f); v1 = fmaxf(v1, 0.f); }
    Cq[rr * ldc + cofs + c0] = v0;
    Cq[rr * ldc + cofs + c1] = v1;
  }
}

// ---------- xr for downscaler: x_high[N,1] @ Wr[1,64] + br ----------
__global__ void k_outer1(const float* __restrict__ xh, const float* __restrict__ Wr,
                         const float* __restrict__ br, float* __restrict__ xr, long n) {
  long i = (long)blockIdx.x * blockDim.x + threadIdx.x;
  long st = (long)gridDim.x * blockDim.x;
  for (; i < n; i += st) {
    long node = i >> 6; int c = (int)(i & 63);
    xr[i] = xh[node] * Wr[c] + br[c];
  }
}

// ---------- degree ----------
__global__ void k_deg(const int* __restrict__ dst, long E, long nself, float* deg) {
  long i = (long)blockIdx.x * blockDim.x + threadIdx.x;
  long st = (long)gridDim.x * blockDim.x;
  long Et = E + nself;
  for (; i < Et; i += st) {
    int d = (i < E) ? dst[i] : (int)(i - E);
    atomicAdd(&deg[d], 1.f);
  }
}

// ---------- P1: logit + segment max (wave per edge) ----------
__global__ void k_edge_logit(const int* __restrict__ src, const int* __restrict__ dst,
                             long E, long nself,
                             const float* __restrict__ xl, const float* __restrict__ xr,
                             const float* __restrict__ att, int H,
                             float* __restrict__ logit, unsigned* __restrict__ mOrd) {
  int lane = threadIdx.x & 31;
  long wid = ((long)blockIdx.x * blockDim.x + threadIdx.x) >> 5;
  long nw  = ((long)gridDim.x * blockDim.x) >> 5;
  long Et = E + nself;
  for (long e = wid; e < Et; e += nw) {
    int s, d;
    if (e < E) { s = src[e]; d = dst[e]; } else { s = (int)(e - E); d = s; }
    const float* pl = xl + (long)s * H * 64;
    const float* pr = xr + (long)d * H * 64;
    for (int h = 0; h < H; ++h) {
      float p0 = 0.f;
#pragma unroll
      for (int c0 = 0; c0 < 64; c0 += 32) {
        int c = c0 + lane;
        float t = pl[h * 64 + c] + pr[h * 64 + c];
        t = (t > 0.f) ? t : 0.2f * t;          // leaky_relu 0.2
        p0 += t * att[h * 64 + c];
      }
      for (int o = 16; o; o >>= 1) p0 += __shfl_xor(p0, o, 32);
      if (lane == 0) {
        logit[e * H + h] = p0;
        atomicMax(&mOrd[(long)d * H + h], encOrder(p0));
      }
    }
  }
}

// ---------- P2: ex = exp(logit - m[dst]); den += ex (in-place on logit) ----------
__global__ void k_edge_exp(const int* __restrict__ dst, long E, long nself, int H,
                           float* __restrict__ logit, const unsigned* __restrict__ mOrd,
                           float* __restrict__ den) {
  long i = (long)blockIdx.x * blockDim.x + threadIdx.x;
  long st = (long)gridDim.x * blockDim.x;
  long Et = E + nself;
  for (; i < Et; i += st) {
    int d = (i < E) ? dst[i] : (int)(i - E);
    for (int h = 0; h < H; ++h) {
      float ex = __expf(logit[i * H + h] - decOrder(mOrd[(long)d * H + h]));
      logit[i * H + h] = ex;
      atomicAdd(&den[(long)d * H + h], ex);
    }
  }
}

// ---------- P3: agg[dst] += alpha * xl[src] (wave per edge) ----------
__global__ void k_edge_agg(const int* __restrict__ src, const int* __restrict__ dst,
                           long E, long nself,
                           const float* __restrict__ xl, const float* __restrict__ ex,
                           const float* __restrict__ den, int H, float* __restrict__ agg) {
  int lane = threadIdx.x & 31;
  long wid = ((long)blockIdx.x * blockDim.x + threadIdx.x) >> 5;
  long nw  = ((long)gridDim.x * blockDim.x) >> 5;
  long Et = E + nself;
  for (long e = wid; e < Et; e += nw) {
    int s, d;
    if (e < E) { s = src[e]; d = dst[e]; } else { s = (int)(e - E); d = s; }
    const float* pl = xl + (long)s * H * 64;
    float* pa = agg + (long)d * H * 64;
    for (int h = 0; h < H; ++h) {
      float al = ex[e * H + h] / fmaxf(den[(long)d * H + h], 1e-16f);
#pragma unroll
      for (int c0 = 0; c0 < 64; c0 += 32) {
        int c = c0 + lane;
        atomicAdd(&pa[h * 64 + c], pl[h * 64 + c] * al);
      }
    }
  }
}

// ---------- P4: out = agg/max(deg,1) + bias (opt relu), into out[ld,cofs] ----------
__global__ void k_gat_out(const float* __restrict__ agg, const float* __restrict__ deg,
                          const float* __restrict__ bias, long N, int HC,
                          float* __restrict__ out, int ldo, int cofs, int relu) {
  long i = (long)blockIdx.x * blockDim.x + threadIdx.x;
  long st = (long)gridDim.x * blockDim.x;
  long tot = N * HC;
  for (; i < tot; i += st) {
    long n = i / HC; int j = (int)(i - n * HC);
    float v = agg[i] / fmaxf(deg[n], 1.f) + bias[j];
    if (relu) v = fmaxf(v, 0.f);
    out[n * ldo + cofs + j] = v;
  }
}

__global__ void k_copy_col(const float* __restrict__ zv, float* __restrict__ out,
                           long N, int ldo) {
  long i = (long)blockIdx.x * blockDim.x + threadIdx.x;
  long st = (long)gridDim.x * blockDim.x;
  for (; i < N; i += st) out[i * ldo] = zv[i];
}

// ---------- BatchNorm ----------
__global__ void k_bn_stats(const float* __restrict__ x, long N, int Cc, int ld,
                           float* __restrict__ sum, float* __restrict__ sumsq) {
  long tid = (long)blockIdx.x * blockDim.x + threadIdx.x;
  long tot = (long)gridDim.x * blockDim.x;
  int col = (int)(tid % Cc);
  long r0 = tid / Cc;
  long rstr = tot / Cc;
  float s = 0.f, q = 0.f;
  for (long r = r0; r < N; r += rstr) {
    float v = x[r * ld + col];
    s += v; q += v * v;
  }
  atomicAdd(&sum[col], s);
  atomicAdd(&sumsq[col], q);
}

__global__ void k_bn_final(const float* __restrict__ sum, const float* __restrict__ sumsq,
                           const float* __restrict__ gamma, const float* __restrict__ beta,
                           long N, int Cc, float* __restrict__ scale, float* __restrict__ shift) {
  int c = blockIdx.x * blockDim.x + threadIdx.x;
  if (c >= Cc) return;
  float inv = 1.f / (float)N;
  float mu = sum[c] * inv;
  float var = sumsq[c] * inv - mu * mu;
  float sc = gamma[c] * rsqrtf(var + 1e-5f);
  scale[c] = sc;
  shift[c] = beta[c] - mu * sc;
}

__global__ void k_bn_apply(float* __restrict__ x, long N, int Cc, int ld,
                           const float* __restrict__ scale, const float* __restrict__ shift,
                           int relu) {
  long i = (long)blockIdx.x * blockDim.x + threadIdx.x;
  long st = (long)gridDim.x * blockDim.x;
  long tot = N * Cc;
  for (; i < tot; i += st) {
    long n = i / Cc; int c = (int)(i - n * Cc);
    float v = x[n * ld + c] * scale[c] + shift[c];
    if (relu) v = fmaxf(v, 0.f);
    x[n * ld + c] = v;
  }
}

// ---------- final head: y2[N,32] @ W3[32,1] + b3 ----------
__global__ void k_pred3(const float* __restrict__ y2, const float* __restrict__ W3,
                        const float* __restrict__ b3, float* __restrict__ out, long N) {
  long i = (long)blockIdx.x * blockDim.x + threadIdx.x;
  long st = (long)gridDim.x * blockDim.x;
  for (; i < N; i += st) {
    const float* r = y2 + i * 32;
    float s = 0.f;
#pragma unroll
    for (int j = 0; j < 32; ++j) s += r[j] * W3[j];
    out[i] = s + b3[0];
  }
}

// ---------- host ----------
static inline unsigned nblk(long n, int bs) {
  long b = (n + bs - 1) / bs;
  if (b < 1) b = 1;
  if (b > 1000000000L) b = 1000000000L;
  return (unsigned)b;
}

extern "C" void kernel_launch(void* const* d_in, const int* in_sizes, int n_in,
                              void* d_out, int out_size, void* d_ws, size_t ws_size,
                              hipStream_t stream) {
  const float* x_low  = (const float*)d_in[0];
  const float* x_high = (const float*)d_in[1];
  const float* z_std  = (const float*)d_in[2];
  const int*   e_l2h  = (const int*)d_in[3];
  const int*   e_hh   = (const int*)d_in[4];
  const float* const* P = (const float* const*)(d_in + 5);
  // P: 0..5 down{Wl,bl,Wr,br,att,bias}; 6,7 bn0; per layer i: 8+(i-1)*8 {Wl,bl,Wr,br,att,bias,gamma,beta}
  // gat5 at 40..45 ; pred at 46..51

  long NL = in_sizes[0] / 125;
  long NH = in_sizes[1];
  long EL = in_sizes[3] / 2;
  long EH = in_sizes[4] / 2;
  const int* l2h_src = e_l2h;
  const int* l2h_dst = e_l2h + EL;
  const int* hh_src = e_hh;
  const int* hh_dst = e_hh + EH;

  // workspace carve-out
  float* bufA = (float*)d_ws;                 // [NH,128]
  float* bufB = bufA + (size_t)NH * 128;      // [NH,128]
  float* XL   = bufB + (size_t)NH * 128;      // [NH,128]
  float* XR   = XL + (size_t)NH * 128;        // [NH,128]
  float* EB   = XR + (size_t)NH * 128;        // edge logits/ex
  size_t Emax = (size_t)(EH + NH) * 2;
  if ((size_t)EL > Emax) Emax = (size_t)EL;
  unsigned* MO = (unsigned*)(EB + Emax);      // [NH*2]
  float* DEN  = (float*)(MO + (size_t)NH * 2);
  float* DEGL = DEN + (size_t)NH * 2;
  float* DEGH = DEGL + NH;
  float* SUM  = DEGH + NH;                    // 128
  float* SUMQ = SUM + 128;
  float* SCL  = SUMQ + 128;
  float* SHF  = SCL + 128;
  _Float16* WT = (_Float16*)(SHF + 128);      // [128,128] f16 = 32KB
  float* PXL = (float*)(WT + 128 * 128);      // padded x_low [NL,128]
  (void)ws_size; (void)n_in; (void)out_size;

#define ZERO(p, n) k_zero<<<nblk((n), 256), 256, 0, stream>>>((p), (long)(n))
#define CONVW(W, K, N, Kp) \
  k_convw<<<nblk((long)(N) * (Kp), 256), 256, 0, stream>>>((W), (K), (N), (Kp), WT)
#define GEMM(A, lda, bias, C, ldc, cofs, M, N, Kp, relu)                                 \
  k_gemm_wmma<<<dim3((unsigned)(((M) + 63) / 64), (unsigned)((N) / 32)), 128, 0,         \
                stream>>>((A), (lda), WT, (bias), (C), (ldc), (cofs), (int)(M), (N),     \
                          (Kp), (relu))

  // degrees (computed once per edge-set)
  ZERO(DEGL, NH);
  k_deg<<<nblk(EL, 256), 256, 0, stream>>>(l2h_dst, EL, 0, DEGL);
  ZERO(DEGH, NH);
  k_deg<<<nblk(EH + NH, 256), 256, 0, stream>>>(hh_dst, EH, NH, DEGH);

  // ===== downscaler (bipartite, H=1) =====
  k_pad<<<nblk(NL * 128, 256), 256, 0, stream>>>(x_low, 125, PXL, 128, NL);
  CONVW(P[0], 125, 64, 128);
  GEMM(PXL, 128, P[1], XL, 64, 0, NL, 64, 128, 0);                     // xl = x_low@Wl+bl
  k_outer1<<<nblk(NH * 64, 256), 256, 0, stream>>>(x_high, P[2], P[3], XR, NH * 64);
  ZERO((float*)MO, NH);
  ZERO(DEN, NH);
  k_edge_logit<<<nblk(EL * 32, 256), 256, 0, stream>>>(l2h_src, l2h_dst, EL, 0, XL, XR,
                                                       P[4], 1, EB, MO);
  k_edge_exp<<<nblk(EL, 256), 256, 0, stream>>>(l2h_dst, EL, 0, 1, EB, MO, DEN);
  ZERO(bufA, NH * 64);
  k_edge_agg<<<nblk(EL * 32, 256), 256, 0, stream>>>(l2h_src, l2h_dst, EL, 0, XL, EB, DEN,
                                                     1, bufA);
  ZERO(bufB, NH * 128);                       // h0: ld=128, cols 65..127 must stay zero
  k_gat_out<<<nblk(NH * 64, 256), 256, 0, stream>>>(bufA, DEGL, P[5], NH, 64, bufB, 128, 1, 0);
  k_copy_col<<<nblk(NH, 256), 256, 0, stream>>>(z_std, bufB, NH, 128);

  // bn0 in-place on bufB (width 65, ld 128, no relu)
  ZERO(SUM, 128); ZERO(SUMQ, 128);
  k_bn_stats<<<65 * 4, 256, 0, stream>>>(bufB, NH, 65, 128, SUM, SUMQ);
  k_bn_final<<<1, 128, 0, stream>>>(SUM, SUMQ, P[6], P[7], NH, 65, SCL, SHF);
  k_bn_apply<<<nblk(NH * 65, 256), 256, 0, stream>>>(bufB, NH, 65, 128, SCL, SHF, 0);

  // ===== processor layers 1..4 (H=2, width -> 128) =====
  float* h = bufB;
  float* t = bufA;
  int win = 65;
  for (int i = 1; i <= 4; ++i) {
    const float* const* Q = P + 8 + (i - 1) * 8;  // Wl,bl,Wr,br,att,bias,gamma,beta
    CONVW(Q[0], win, 128, 128);
    GEMM(h, 128, Q[1], XL, 128, 0, NH, 128, 128, 0);
    CONVW(Q[2], win, 128, 128);
    GEMM(h, 128, Q[3], XR, 128, 0, NH, 128, 128, 0);
    ZERO((float*)MO, NH * 2);
    ZERO(DEN, NH * 2);
    k_edge_logit<<<nblk((EH + NH) * 32, 256), 256, 0, stream>>>(hh_src, hh_dst, EH, NH, XL,
                                                                XR, Q[4], 2, EB, MO);
    k_edge_exp<<<nblk(EH + NH, 256), 256, 0, stream>>>(hh_dst, EH, NH, 2, EB, MO, DEN);
    ZERO(t, NH * 128);
    k_edge_agg<<<nblk((EH + NH) * 32, 256), 256, 0, stream>>>(hh_src, hh_dst, EH, NH, XL,
                                                              EB, DEN, 2, t);
    k_gat_out<<<nblk(NH * 128, 256), 256, 0, stream>>>(t, DEGH, Q[5], NH, 128, t, 128, 0, 0);
    ZERO(SUM, 128); ZERO(SUMQ, 128);
    k_bn_stats<<<128 * 4, 256, 0, stream>>>(t, NH, 128, 128, SUM, SUMQ);
    k_bn_final<<<1, 128, 0, stream>>>(SUM, SUMQ, Q[6], Q[7], NH, 128, SCL, SHF);
    k_bn_apply<<<nblk(NH * 128, 256), 256, 0, stream>>>(t, NH, 128, 128, SCL, SHF, 1);
    float* tmp = h; h = t; t = tmp;
    win = 128;
  }

  // ===== gat5 (H=1, 128 -> 64) + relu =====
  CONVW(P[40], 128, 64, 128);
  GEMM(h, 128, P[41], XL, 64, 0, NH, 64, 128, 0);
  CONVW(P[42], 128, 64, 128);
  GEMM(h, 128, P[43], XR, 64, 0, NH, 64, 128, 0);
  ZERO((float*)MO, NH);
  ZERO(DEN, NH);
  k_edge_logit<<<nblk((EH + NH) * 32, 256), 256, 0, stream>>>(hh_src, hh_dst, EH, NH, XL,
                                                              XR, P[44], 1, EB, MO);
  k_edge_exp<<<nblk(EH + NH, 256), 256, 0, stream>>>(hh_dst, EH, NH, 1, EB, MO, DEN);
  ZERO(t, NH * 64);
  k_edge_agg<<<nblk((EH + NH) * 32, 256), 256, 0, stream>>>(hh_src, hh_dst, EH, NH, XL, EB,
                                                            DEN, 1, t);
  k_gat_out<<<nblk(NH * 64, 256), 256, 0, stream>>>(t, DEGH, P[45], NH, 64, t, 64, 0, 1);

  // ===== predictor MLP =====
  CONVW(P[46], 64, 64, 64);
  GEMM(t, 64, P[47], XL, 64, 0, NH, 64, 64, 1);    // y1 = relu(h@W1+b1)  [NH,64]
  CONVW(P[48], 64, 32, 64);
  GEMM(XL, 64, P[49], XR, 32, 0, NH, 32, 64, 1);   // y2 = relu(y1@W2+b2) [NH,32]
  k_pred3<<<nblk(NH, 256), 256, 0, stream>>>(XR, P[50], P[51], (float*)d_out, NH);

#undef ZERO
#undef CONVW
#undef GEMM
}
